// Transformer_50809463111778
// MI455X (gfx1250) — matrix operand order repaired and checked by reading, not compiled, not run on hardware
//
#include <hip/hip_runtime.h>
#include <hip/hip_bf16.h>
#include <math.h>

// ---------------------------------------------------------------------------
// Problem constants (match reference)
// ---------------------------------------------------------------------------
#define BATCH  64
#define SEQL   96
#define NTOK   (BATCH * SEQL)   // 6144
#define DIM    256
#define NHEAD  8
#define DKH    32
#define NLAYER 2
#define VOCAB  1024
#define INV_SCALE 0.17677669529663687f   // 1/sqrt(32)

typedef __bf16 bf16;
typedef __attribute__((ext_vector_type(16))) __bf16 v16bf;
typedef __attribute__((ext_vector_type(8)))  __bf16 v8bf;
typedef __attribute__((ext_vector_type(8)))  float  v8f;
typedef __attribute__((ext_vector_type(4)))  int    v4i;

#define AS_GLOBAL __attribute__((address_space(1)))
#define AS_LDS    __attribute__((address_space(3)))

#if __has_builtin(__builtin_amdgcn_global_load_async_to_lds_b128)
#define HAVE_ASYNC_LDS 1
#endif

__device__ __forceinline__ void wait_async0() {
#if __has_builtin(__builtin_amdgcn_s_wait_asynccnt)
    __builtin_amdgcn_s_wait_asynccnt(0);
#else
    asm volatile("s_wait_asynccnt 0" ::: "memory");
#endif
}

// Pack two fp32 into packed bf16 pair (lo = a, hi = b), round-to-nearest via
// +0x8000 then take high halves with a single v_perm_b32.
__device__ __forceinline__ unsigned pk2bf(float a, float b) {
    unsigned ua = __builtin_bit_cast(unsigned, a) + 0x8000u;
    unsigned ub = __builtin_bit_cast(unsigned, b) + 0x8000u;
    // D[15:0] = ua[31:16] (sel bytes 2,3 of src1), D[31:16] = ub[31:16] (bytes 6,7 of src0)
    return __builtin_amdgcn_perm(ub, ua, 0x07060302u);
}

// ---------------------------------------------------------------------------
// Embedding: x[n][d] = coord_emb[pos%3][d] + pos_emb[pos/3][d] + value_emb[tok][d]
// ---------------------------------------------------------------------------
__global__ __launch_bounds__(256) void embed_kernel(
    const int* __restrict__ tok, const int* __restrict__ pos,
    const float* __restrict__ vemb, const float* __restrict__ cemb,
    const float* __restrict__ pemb, float* __restrict__ x)
{
    int row = blockIdx.x;
    int col = threadIdx.x;
    int p = pos[row];
    x[(size_t)row * DIM + col] =
        cemb[(p % 3) * DIM + col] + pemb[(p / 3) * DIM + col] +
        vemb[(size_t)tok[row] * DIM + col];
}

// ---------------------------------------------------------------------------
// LayerNorm over D=256, one block per row.
// ---------------------------------------------------------------------------
__global__ __launch_bounds__(256) void ln_kernel(
    const float* __restrict__ x, const float* __restrict__ s,
    const float* __restrict__ b, float* __restrict__ y)
{
    __shared__ float red[256];
    int row = blockIdx.x, t = threadIdx.x;
    float v = x[(size_t)row * DIM + t];
    red[t] = v; __syncthreads();
    for (int o = 128; o; o >>= 1) { if (t < o) red[t] += red[t + o]; __syncthreads(); }
    float mu = red[0] * (1.0f / DIM);
    __syncthreads();
    float d = v - mu;
    red[t] = d * d; __syncthreads();
    for (int o = 128; o; o >>= 1) { if (t < o) red[t] += red[t + o]; __syncthreads(); }
    float var = red[0] * (1.0f / DIM);
    y[(size_t)row * DIM + t] = d * rsqrtf(var + 1e-5f) * s[t] + b[t];
}

// ---------------------------------------------------------------------------
// WMMA GEMM:  C[M x Ncol] = act( A[M x K] * W[K x Ncol] + bias ) (+ res)
// 64x64 block tile, 256 threads = 8 waves; wave (wm,wn) owns a 16x32 subtile.
// fp32 inputs are rounded+packed to bf16 pairs with v_perm_b32 during staging:
//   A: one ds_store_b128 per thread per tile (8 elems, K-contiguous)
//   B: four ds_store_b32 per thread (K-pairs down a column of Bs[n][k])
// Fragment layouts per CDNA5 ISA (wave32).
// ---------------------------------------------------------------------------
#define BM 64
#define BN 64
#define BK 32
#define LPAD 16   // LDS row pad (bf16 elems); row stride 96B, 16B-aligned

__global__ __launch_bounds__(256) void gemm_wmma_kernel(
    const float* __restrict__ A, const float* __restrict__ W,
    const float* __restrict__ bias, const float* __restrict__ res,
    float* __restrict__ C, int K, int Ncol, int relu)
{
    __shared__ alignas(16) bf16 As[BM][BK + LPAD];
    __shared__ alignas(16) bf16 Bs[BN][BK + LPAD];   // transposed: Bs[n][k]

    const int t    = threadIdx.x;
    const int m0   = blockIdx.y * BM;
    const int n0   = blockIdx.x * BN;
    const int lane = t & 31;
    const int w    = t >> 5;
    const int wm   = w & 3;          // 4 row tiles of 16
    const int wn   = w >> 2;         // 2 col tiles of 32
    const int half = lane >> 4;
    const int l16  = lane & 15;

    // staging assignments
    const int ar  = t >> 2;          // 0..63  (A row in tile)
    const int ac  = (t & 3) * 8;     // 0,8,16,24 (A col in tile)
    const int bk2 = (t >> 4) * 2;    // 0,2,..,30 (even K in tile)
    const int bn4 = (t & 15) * 4;    // 0,4,..,60 (W col in tile)

    v8f acc0 = {}; v8f acc1 = {};

    for (int kt = 0; kt < K; kt += BK) {
        // ---- stage A tile: 2x float4 loads -> 4 packed dwords -> 1 b128 store
        const float4 a0 = *(const float4*)&A[(size_t)(m0 + ar) * K + kt + ac];
        const float4 a1 = *(const float4*)&A[(size_t)(m0 + ar) * K + kt + ac + 4];
        uint4 apk;
        apk.x = pk2bf(a0.x, a0.y);  apk.y = pk2bf(a0.z, a0.w);
        apk.z = pk2bf(a1.x, a1.y);  apk.w = pk2bf(a1.z, a1.w);
        *(uint4*)&As[ar][ac] = apk;

        // ---- stage B tile transposed: K-pair packed per column cell
        const float4 w0 = *(const float4*)&W[(size_t)(kt + bk2) * Ncol + n0 + bn4];
        const float4 w1 = *(const float4*)&W[(size_t)(kt + bk2 + 1) * Ncol + n0 + bn4];
        *(unsigned*)&Bs[bn4 + 0][bk2] = pk2bf(w0.x, w1.x);
        *(unsigned*)&Bs[bn4 + 1][bk2] = pk2bf(w0.y, w1.y);
        *(unsigned*)&Bs[bn4 + 2][bk2] = pk2bf(w0.z, w1.z);
        *(unsigned*)&Bs[bn4 + 3][bk2] = pk2bf(w0.w, w1.w);
        __syncthreads();

        if (kt + BK < K) {   // prefetch next tiles -> global_prefetch_b8
            __builtin_prefetch(&A[(size_t)(m0 + ar) * K + kt + BK + ac], 0, 0);
            __builtin_prefetch(&W[(size_t)(kt + BK + bk2) * Ncol + n0 + bn4], 0, 0);
        }

        // ---- A fragment: lane m=l16, half selects K +/- 8, two b128 ds loads
        const int am = wm * 16 + l16;
        v8bf alo = *(const v8bf*)&As[am][half * 8];
        v8bf ahi = *(const v8bf*)&As[am][16 + half * 8];
        v16bf afrag = __builtin_shufflevector(alo, ahi,
            0,1,2,3,4,5,6,7,8,9,10,11,12,13,14,15);

        // ---- B fragments (two 16-col tiles): half selects K +/- 8, same as A
        const int nb0 = wn * 32 + l16;
        v8bf b0lo = *(const v8bf*)&Bs[nb0][half * 8];
        v8bf b0hi = *(const v8bf*)&Bs[nb0][16 + half * 8];
        v16bf bfrag0 = __builtin_shufflevector(b0lo, b0hi,
            0,1,2,3,4,5,6,7,8,9,10,11,12,13,14,15);
        const int nb1 = nb0 + 16;
        v8bf b1lo = *(const v8bf*)&Bs[nb1][half * 8];
        v8bf b1hi = *(const v8bf*)&Bs[nb1][16 + half * 8];
        v16bf bfrag1 = __builtin_shufflevector(b1lo, b1hi,
            0,1,2,3,4,5,6,7,8,9,10,11,12,13,14,15);

        acc0 = __builtin_amdgcn_wmma_f32_16x16x32_bf16(
            false, afrag, false, bfrag0, (short)0, acc0, false, false);
        acc1 = __builtin_amdgcn_wmma_f32_16x16x32_bf16(
            false, afrag, false, bfrag1, (short)0, acc1, false, false);
        __syncthreads();
    }

    // ---- epilogue: C layout lane n=lane%16; VGPR r -> M = half*8 + r ----
    const int row0 = m0 + wm * 16 + half * 8;
    const int c0   = n0 + wn * 32 + l16;
    const int c1   = c0 + 16;
    const float bia0 = bias[c0], bia1 = bias[c1];
    #pragma unroll
    for (int r = 0; r < 8; ++r) {
        int row = row0 + r;
        float v0 = acc0[r] + bia0;
        float v1 = acc1[r] + bia1;
        if (relu) { v0 = fmaxf(v0, 0.0f); v1 = fmaxf(v1, 0.0f); }
        if (res) {
            v0 += res[(size_t)row * Ncol + c0];
            v1 += res[(size_t)row * Ncol + c1];
        }
        C[(size_t)row * Ncol + c0] = v0;
        C[(size_t)row * Ncol + c1] = v1;
    }
}

// ---------------------------------------------------------------------------
// Causal attention with exp(clip(qk/sqrt(dk))) weights.
// One block per (batch, head). K/V head slices staged to LDS via
// GLOBAL_LOAD_ASYNC_TO_LDS_B128 (ASYNCcnt) when the builtin exists.
// ---------------------------------------------------------------------------
__global__ __launch_bounds__(256) void attn_kernel(
    const float* __restrict__ qkv, float* __restrict__ o)
{
    __shared__ alignas(16) float Ks[SEQL][DKH];   // row stride 128B, 16B aligned
    __shared__ alignas(16) float Vs[SEQL][DKH];
    const int bh = blockIdx.x;
    const int b  = bh >> 3;
    const int h  = bh & 7;
    const int t  = threadIdx.x;

#if defined(HAVE_ASYNC_LDS)
    for (int e = t; e < SEQL * DKH / 4; e += 256) {     // 768 float4 quads
        int r = e >> 3, c4 = (e & 7) * 4;
        size_t base = (size_t)(b * SEQL + r) * (3 * DIM) + h * DKH + c4;
        __builtin_amdgcn_global_load_async_to_lds_b128(
            (AS_GLOBAL v4i*)const_cast<float*>(qkv + base + DIM),
            (AS_LDS v4i*)&Ks[r][c4], 0, 0);
        __builtin_amdgcn_global_load_async_to_lds_b128(
            (AS_GLOBAL v4i*)const_cast<float*>(qkv + base + 2 * DIM),
            (AS_LDS v4i*)&Vs[r][c4], 0, 0);
    }
    wait_async0();
#else
    for (int e = t; e < SEQL * DKH; e += 256) {
        int r = e >> 5, c = e & 31;
        size_t base = (size_t)(b * SEQL + r) * (3 * DIM) + h * DKH + c;
        Ks[r][c] = qkv[base + DIM];
        Vs[r][c] = qkv[base + 2 * DIM];
    }
#endif
    __syncthreads();

    const int w = t >> 5, lane = t & 31;
    for (int i = w; i < SEQL; i += 8) {
        float q = qkv[(size_t)(b * SEQL + i) * (3 * DIM) + h * DKH + lane];
        float wv = 0.0f, z = 0.0f;
        for (int j = 0; j <= i; ++j) {
            float s = q * Ks[j][lane];
            #pragma unroll
            for (int off = 16; off; off >>= 1) s += __shfl_xor(s, off, 32);
            s *= INV_SCALE;
            s = fminf(fmaxf(s, -5.0f), 5.0f);
            float e = expf(s);
            wv = fmaf(e, Vs[j][lane], wv);
            z += e;
        }
        o[(size_t)(b * SEQL + i) * DIM + h * DKH + lane] = wv / z;
    }
}

// ---------------------------------------------------------------------------
// In-place log_softmax over VOCAB=1024, one block per row, 4 elems/thread.
// ---------------------------------------------------------------------------
__global__ __launch_bounds__(256) void lsm_kernel(float* __restrict__ y)
{
    __shared__ float red[256];
    const int row = blockIdx.x, t = threadIdx.x;
    float v[4];
    #pragma unroll
    for (int k = 0; k < 4; ++k) v[k] = y[(size_t)row * VOCAB + k * 256 + t];
    float m = fmaxf(fmaxf(v[0], v[1]), fmaxf(v[2], v[3]));
    red[t] = m; __syncthreads();
    for (int o = 128; o; o >>= 1) { if (t < o) red[t] = fmaxf(red[t], red[t + o]); __syncthreads(); }
    m = red[0]; __syncthreads();
    float s = 0.0f;
    #pragma unroll
    for (int k = 0; k < 4; ++k) s += expf(v[k] - m);
    red[t] = s; __syncthreads();
    for (int o = 128; o; o >>= 1) { if (t < o) red[t] += red[t + o]; __syncthreads(); }
    float ls = logf(red[0]) + m;
    #pragma unroll
    for (int k = 0; k < 4; ++k) y[(size_t)row * VOCAB + k * 256 + t] = v[k] - ls;
}

// ---------------------------------------------------------------------------
// Host-side orchestration
// ---------------------------------------------------------------------------
extern "C" void kernel_launch(void* const* d_in, const int* in_sizes, int n_in,
                              void* d_out, int out_size, void* d_ws, size_t ws_size,
                              hipStream_t stream)
{
    (void)in_sizes; (void)n_in; (void)out_size; (void)ws_size;

    const int*   tokens = (const int*)  d_in[0];
    const int*   pos    = (const int*)  d_in[1];
    // d_in[2]/d_in[3]: edge_src/edge_dst — equivalent to dense causal mask, unused
    const float* vemb   = (const float*)d_in[4];
    const float* cemb   = (const float*)d_in[5];
    const float* pemb   = (const float*)d_in[6];
    const float* ln1_s  = (const float*)d_in[7];
    const float* ln1_b  = (const float*)d_in[8];
    const float* W_qkv  = (const float*)d_in[9];
    const float* b_qkv  = (const float*)d_in[10];
    const float* W_o    = (const float*)d_in[11];
    const float* b_o    = (const float*)d_in[12];
    const float* ln2_s  = (const float*)d_in[13];
    const float* ln2_b  = (const float*)d_in[14];
    const float* W_ff1  = (const float*)d_in[15];
    const float* b_ff1  = (const float*)d_in[16];
    const float* W_ff2  = (const float*)d_in[17];
    const float* b_ff2  = (const float*)d_in[18];
    const float* W_gen  = (const float*)d_in[19];
    const float* b_gen  = (const float*)d_in[20];

    float* x    = (float*)d_ws;                       // [NTOK x 256]
    float* xn   = x    + (size_t)NTOK * DIM;          // [NTOK x 256] (LN out / attn out)
    float* qkvb = xn   + (size_t)NTOK * DIM;          // [NTOK x 768]
    float* h1   = qkvb + (size_t)NTOK * 3 * DIM;      // [NTOK x 1024]
    float* out  = (float*)d_out;                      // [NTOK x 1024]

    embed_kernel<<<NTOK, 256, 0, stream>>>(tokens, pos, vemb, cemb, pemb, x);

    const int MB = NTOK / BM;   // 96 row blocks
    for (int l = 0; l < NLAYER; ++l) {
        ln_kernel<<<NTOK, 256, 0, stream>>>(x, ln1_s + l * DIM, ln1_b + l * DIM, xn);
        gemm_wmma_kernel<<<dim3(3 * DIM / BN, MB), 256, 0, stream>>>(
            xn, W_qkv + (size_t)l * DIM * 3 * DIM, b_qkv + l * 3 * DIM,
            nullptr, qkvb, DIM, 3 * DIM, 0);
        attn_kernel<<<BATCH * NHEAD, 256, 0, stream>>>(qkvb, xn);
        gemm_wmma_kernel<<<dim3(DIM / BN, MB), 256, 0, stream>>>(
            xn, W_o + (size_t)l * DIM * DIM, b_o + l * DIM,
            x, x, DIM, DIM, 0);
        ln_kernel<<<NTOK, 256, 0, stream>>>(x, ln2_s + l * DIM, ln2_b + l * DIM, xn);
        gemm_wmma_kernel<<<dim3(4 * DIM / BN, MB), 256, 0, stream>>>(
            xn, W_ff1 + (size_t)l * DIM * 4 * DIM, b_ff1 + l * 4 * DIM,
            nullptr, h1, DIM, 4 * DIM, 1);
        gemm_wmma_kernel<<<dim3(DIM / BN, MB), 256, 0, stream>>>(
            h1, W_ff2 + (size_t)l * 4 * DIM * DIM, b_ff2 + l * DIM,
            x, x, 4 * DIM, DIM, 0);
    }

    gemm_wmma_kernel<<<dim3(VOCAB / BN, MB), 256, 0, stream>>>(
        x, W_gen, b_gen, nullptr, out, DIM, VOCAB, 0);
    lsm_kernel<<<NTOK, 256, 0, stream>>>(out);
}
